// Retina_47004122087473
// MI455X (gfx1250) — matrix operand order repaired and checked by reading, not compiled, not run on hardware
//
#include <hip/hip_runtime.h>
#include <math.h>

// ---------------------------------------------------------------------------
// Types for CDNA5 WMMA (wave32): v_wmma_f32_16x16x32_bf16
// ---------------------------------------------------------------------------
typedef __attribute__((ext_vector_type(16))) __bf16 v16bf;
typedef __attribute__((ext_vector_type(8)))  __bf16 v8bf;
typedef __attribute__((ext_vector_type(8)))  float  v8f;

union V16 { v16bf v; v8bf h[2]; };

__device__ __forceinline__ v8bf zero8() {
    v8bf z;
#pragma unroll
    for (int i = 0; i < 8; ++i) z[i] = (__bf16)0.0f;
    return z;
}

// bilinear blend of four 8-wide bf16 chunks -> bf16 chunk
__device__ __forceinline__ v8bf bilin8(const __bf16* p00, const __bf16* p01,
                                       const __bf16* p10, const __bf16* p11,
                                       float fx, float fy) {
    v8bf a = *(const v8bf*)p00;
    v8bf b = *(const v8bf*)p01;
    v8bf c = *(const v8bf*)p10;
    v8bf d = *(const v8bf*)p11;
    v8bf r;
#pragma unroll
    for (int i = 0; i < 8; ++i) {
        float top = (1.0f - fx) * (float)a[i] + fx * (float)b[i];
        float bot = (1.0f - fx) * (float)c[i] + fx * (float)d[i];
        r[i] = (__bf16)((1.0f - fy) * top + fy * bot);
    }
    return r;
}

// ---------------------------------------------------------------------------
// Prep kernels: dtype/layout conversion
// ---------------------------------------------------------------------------
// fuse_w: fp32 [256][1024][1][1] -> bf16 [256][1024] (same order)
__global__ void cvt_flat_bf16(const float* __restrict__ in, __bf16* __restrict__ out, int n) {
    int i = blockIdx.x * 256 + threadIdx.x;
    if (i < n) out[i] = (__bf16)in[i];
}

// 3x3 weights: fp32 [CO_in][256][3][3] -> bf16 [CO_pad][9][256], zero-pad rows
__global__ void cvt_w3x3_bf16(const float* __restrict__ in, __bf16* __restrict__ out,
                              int CO_in, int CO_pad) {
    size_t i = (size_t)blockIdx.x * 256 + threadIdx.x;
    size_t total = (size_t)CO_pad * 9 * 256;
    if (i >= total) return;
    int ci = (int)(i & 255);
    int rs = (int)((i >> 8) % 9);
    int co = (int)(i / (9 * 256));
    out[i] = (co < CO_in) ? (__bf16)in[((size_t)co * 256 + ci) * 9 + rs] : (__bf16)0.0f;
}

// feature maps: fp32 NCHW [16][256][HW] -> bf16 NHWC [16][HW][256]
__global__ void cvt_nchw_nhwc(const float* __restrict__ in, __bf16* __restrict__ out, int HW) {
    size_t i = (size_t)blockIdx.x * 256 + threadIdx.x;
    size_t total = (size_t)16 * HW * 256;
    if (i >= total) return;
    int c = (int)(i & 255);
    size_t p = i >> 8;                 // n*HW + hw
    int hw = (int)(p % HW);
    int n  = (int)(p / HW);
    out[i] = (__bf16)in[((size_t)n * 256 + c) * HW + hw];
}

// ---------------------------------------------------------------------------
// Fuse kernel: 1x1 conv over concat(fm0, up(fm1), up(fm2), up(fm3)), K=1024.
// Implicit GEMM with on-the-fly bilinear sampling of native-res bf16 levels.
// Block = 8 waves (4 co-tiles x 2 w-tiles); wave tile = 16 co x 16 w.
// ---------------------------------------------------------------------------
__global__ __launch_bounds__(256) void fuse_wmma(
    const __bf16* __restrict__ L0, const __bf16* __restrict__ L1,
    const __bf16* __restrict__ L2, const __bf16* __restrict__ L3,
    const __bf16* __restrict__ WF,          // [256][1024] bf16
    const float*  __restrict__ BIAS,        // [256] fp32
    __bf16* __restrict__ Y)                 // [16][56][56][256] bf16
{
    const int lane = threadIdx.x & 31;
    const int wv   = threadIdx.x >> 5;
    const int mi = wv & 3, ni = wv >> 2;
    const int cot = blockIdx.x >> 1, wt = blockIdx.x & 1;
    const int h = blockIdx.y, n = blockIdx.z;
    const int co_w = cot * 64 + mi * 16;
    const int lcol = lane & 15;
    const bool hi = lane >= 16;
    const int wcol = wt * 32 + ni * 16 + lcol;
    const int wld  = (wcol < 56) ? wcol : 55;

    const __bf16* wrow = WF + (size_t)(co_w + lcol) * 1024;

    v8f acc = {};

    // level 0: direct NHWC load
    {
        const __bf16* x0 = L0 + (((size_t)n * 56 + h) * 56 + wld) * 256;
#pragma unroll
        for (int kc = 0; kc < 256; kc += 32) {
            V16 a, b;
            const __bf16* wp = wrow + kc + (hi ? 8 : 0);
            a.h[0] = *(const v8bf*)(wp);
            a.h[1] = *(const v8bf*)(wp + 16);
            const __bf16* xp = x0 + kc + (hi ? 8 : 0);
            b.h[0] = *(const v8bf*)(xp);
            b.h[1] = *(const v8bf*)(xp + 16);
            acc = __builtin_amdgcn_wmma_f32_16x16x32_bf16(
                false, a.v, false, b.v, (short)0, acc, false, false);
        }
    }

    // levels 1..3: bilinear (half-pixel centers, clamp-to-edge)
#pragma unroll
    for (int l = 1; l < 4; ++l) {
        const int Hl = 56 >> l;
        const __bf16* base =
            ((l == 1) ? L1 : (l == 2) ? L2 : L3) + (size_t)n * Hl * Hl * 256;
        const float sc = 1.0f / (float)(1 << l);
        float sy = (h + 0.5f) * sc - 0.5f;
        int   y0 = (int)floorf(sy);
        float fy = sy - (float)y0;
        int y0c = y0 < 0 ? 0 : (y0 > Hl - 1 ? Hl - 1 : y0);
        int y1c = (y0 + 1) < 0 ? 0 : ((y0 + 1) > Hl - 1 ? Hl - 1 : (y0 + 1));
        float sx = (wld + 0.5f) * sc - 0.5f;
        int   x0 = (int)floorf(sx);
        float fx = sx - (float)x0;
        int x0c = x0 < 0 ? 0 : (x0 > Hl - 1 ? Hl - 1 : x0);
        int x1c = (x0 + 1) < 0 ? 0 : ((x0 + 1) > Hl - 1 ? Hl - 1 : (x0 + 1));
        const __bf16* p00 = base + ((size_t)y0c * Hl + x0c) * 256;
        const __bf16* p01 = base + ((size_t)y0c * Hl + x1c) * 256;
        const __bf16* p10 = base + ((size_t)y1c * Hl + x0c) * 256;
        const __bf16* p11 = base + ((size_t)y1c * Hl + x1c) * 256;

#pragma unroll
        for (int kc = 0; kc < 256; kc += 32) {
            const int off = kc + (hi ? 8 : 0);
            V16 a, b;
            b.h[0] = bilin8(p00 + off,     p01 + off,     p10 + off,     p11 + off,     fx, fy);
            b.h[1] = bilin8(p00 + off + 16, p01 + off + 16, p10 + off + 16, p11 + off + 16, fx, fy);
            const __bf16* wp = wrow + l * 256 + kc + (hi ? 8 : 0);
            a.h[0] = *(const v8bf*)(wp);
            a.h[1] = *(const v8bf*)(wp + 16);
            acc = __builtin_amdgcn_wmma_f32_16x16x32_bf16(
                false, a.v, false, b.v, (short)0, acc, false, false);
        }
    }

    // epilogue: + bias (no ReLU), store bf16 NHWC
    if (wcol < 56) {
        const int mbase = hi ? 8 : 0;
        __bf16* yp = Y + (((size_t)n * 56 + h) * 56 + wcol) * 256 + co_w + mbase;
        v8bf o;
#pragma unroll
        for (int i = 0; i < 8; ++i)
            o[i] = (__bf16)(acc[i] + BIAS[co_w + mbase + i]);
        *(v8bf*)yp = o;
    }
}

// ---------------------------------------------------------------------------
// 3x3 conv 256->256 + bias + ReLU (subnet layer), bf16 NHWC in/out.
// K = 9 * 256; rs-outer / ci-inner ordering matches weight layout [CO][9][256].
// Register-blocked 2x2: each wave owns 4 accumulator tiles (32co x 32w), so
// every A and B fragment feeds two WMMAs -> 2 b128 loads per WMMA instead of 4.
// Block = 8 waves = 4(M) x 2(N) -> 128co x 64w per block; grid (2, 56, 16).
// ---------------------------------------------------------------------------
__global__ __launch_bounds__(256) void conv3x3_wmma(
    const __bf16* __restrict__ X,   // [16][56][56][256]
    const __bf16* __restrict__ W,   // [256][9][256]
    const float*  __restrict__ BIAS,// [256]
    __bf16* __restrict__ Y)         // [16][56][56][256]
{
    const int lane = threadIdx.x & 31;
    const int wv   = threadIdx.x >> 5;
    const int mi = wv & 3, ni = wv >> 2;
    const int cot = blockIdx.x;                 // 0..1
    const int h = blockIdx.y, n = blockIdx.z;
    const int co_w = cot * 128 + mi * 32;       // wave: co_w .. co_w+31
    const int lcol = lane & 15;
    const bool hi = lane >= 16;
    const int wcol0 = ni * 32 + lcol;           // wave: columns [ni*32, ni*32+31]
    const int wcol1 = wcol0 + 16;

    const __bf16* wbase0 = W + (size_t)(co_w + lcol) * 2304;
    const __bf16* wbase1 = wbase0 + (size_t)16 * 2304;

    v8f acc00 = {}, acc01 = {}, acc10 = {}, acc11 = {};

    for (int rs = 0; rs < 9; ++rs) {
        const int r = rs / 3, s = rs - r * 3;
        const int hh = h + r - 1;
        if (hh < 0 || hh >= 56) continue;       // uniform per block
        const int wc0 = wcol0 + s - 1;
        const int wc1 = wcol1 + s - 1;
        const bool bv0 = (wc0 >= 0 && wc0 < 56);
        const bool bv1 = (wc1 >= 0 && wc1 < 56);
        const __bf16* xr = X + (((size_t)n * 56 + hh) * 56) * 256;
        const __bf16* xrow0 = xr + (size_t)(bv0 ? wc0 : 0) * 256;
        const __bf16* xrow1 = xr + (size_t)(bv1 ? wc1 : 0) * 256;
        const __bf16* wrow0 = wbase0 + rs * 256;
        const __bf16* wrow1 = wbase1 + rs * 256;
#pragma unroll
        for (int kc = 0; kc < 256; kc += 32) {
            V16 a0, a1, b0, b1;
            const __bf16* wp0 = wrow0 + kc + (hi ? 8 : 0);
            const __bf16* wp1 = wrow1 + kc + (hi ? 8 : 0);
            a0.h[0] = *(const v8bf*)(wp0);
            a0.h[1] = *(const v8bf*)(wp0 + 16);
            a1.h[0] = *(const v8bf*)(wp1);
            a1.h[1] = *(const v8bf*)(wp1 + 16);
            if (bv0) {
                const __bf16* xp = xrow0 + kc + (hi ? 8 : 0);
                b0.h[0] = *(const v8bf*)(xp);
                b0.h[1] = *(const v8bf*)(xp + 16);
            } else { b0.h[0] = zero8(); b0.h[1] = zero8(); }
            if (bv1) {
                const __bf16* xp = xrow1 + kc + (hi ? 8 : 0);
                b1.h[0] = *(const v8bf*)(xp);
                b1.h[1] = *(const v8bf*)(xp + 16);
            } else { b1.h[0] = zero8(); b1.h[1] = zero8(); }
            acc00 = __builtin_amdgcn_wmma_f32_16x16x32_bf16(
                false, a0.v, false, b0.v, (short)0, acc00, false, false);
            acc01 = __builtin_amdgcn_wmma_f32_16x16x32_bf16(
                false, a0.v, false, b1.v, (short)0, acc01, false, false);
            acc10 = __builtin_amdgcn_wmma_f32_16x16x32_bf16(
                false, a1.v, false, b0.v, (short)0, acc10, false, false);
            acc11 = __builtin_amdgcn_wmma_f32_16x16x32_bf16(
                false, a1.v, false, b1.v, (short)0, acc11, false, false);
        }
    }

    const int mbase = hi ? 8 : 0;
    const size_t rowbase = (((size_t)n * 56 + h) * 56) * 256;

    auto store_tile = [&](const v8f& acc, int coT, int wcT) {
        if (wcT < 56) {
            __bf16* yp = Y + rowbase + (size_t)wcT * 256 + coT + mbase;
            v8bf o;
#pragma unroll
            for (int i = 0; i < 8; ++i) {
                float v = acc[i] + BIAS[coT + mbase + i];
                o[i] = (__bf16)(v > 0.0f ? v : 0.0f);
            }
            *(v8bf*)yp = o;
        }
    };
    store_tile(acc00, co_w,      wcol0);
    store_tile(acc01, co_w,      wcol1);
    store_tile(acc10, co_w + 16, wcol0);
    store_tile(acc11, co_w + 16, wcol1);
}

// ---------------------------------------------------------------------------
// 3x3 head conv 256->CO (90 or 36), bias, fp32 output in permuted layout
// out[n][(h*56+w)][c] with per-image stride 3136*CO (== the A*K permute).
// ---------------------------------------------------------------------------
__global__ __launch_bounds__(256) void head3x3_wmma(
    const __bf16* __restrict__ X,   // [16][56][56][256]
    const __bf16* __restrict__ W,   // [CO_pad][9][256]
    const float*  __restrict__ BIAS,// [CO]
    float* __restrict__ Y,          // [16][3136][CO]
    int CO, int CO_pad)
{
    const int lane = threadIdx.x & 31;
    const int wv   = threadIdx.x >> 5;
    const int mi = wv & 3, ni = wv >> 2;
    const int cot = blockIdx.x >> 1, wt = blockIdx.x & 1;
    const int h = blockIdx.y, n = blockIdx.z;
    const int co_w = cot * 64 + mi * 16;
    if (co_w >= CO_pad) return;                     // whole-wave skip
    const int lcol = lane & 15;
    const bool hi = lane >= 16;
    const int wcol = wt * 32 + ni * 16 + lcol;

    const __bf16* wbase = W + (size_t)(co_w + lcol) * (9 * 256);

    v8f acc = {};
    for (int rs = 0; rs < 9; ++rs) {
        const int r = rs / 3, s = rs - r * 3;
        const int hh = h + r - 1;
        if (hh < 0 || hh >= 56) continue;
        const int wc = wcol + s - 1;
        const bool bval = (wc >= 0 && wc < 56);
        const __bf16* xrow = X + (((size_t)n * 56 + hh) * 56 + (bval ? wc : 0)) * 256;
        const __bf16* wrow = wbase + rs * 256;
#pragma unroll
        for (int kc = 0; kc < 256; kc += 32) {
            V16 a, b;
            const __bf16* wp = wrow + kc + (hi ? 8 : 0);
            a.h[0] = *(const v8bf*)(wp);
            a.h[1] = *(const v8bf*)(wp + 16);
            if (bval) {
                const __bf16* xp = xrow + kc + (hi ? 8 : 0);
                b.h[0] = *(const v8bf*)(xp);
                b.h[1] = *(const v8bf*)(xp + 16);
            } else {
                b.h[0] = zero8();
                b.h[1] = zero8();
            }
            acc = __builtin_amdgcn_wmma_f32_16x16x32_bf16(
                false, a.v, false, b.v, (short)0, acc, false, false);
        }
    }

    if (wcol < 56) {
        const int mbase = hi ? 8 : 0;
        float* yb = Y + (size_t)n * 3136 * CO + (size_t)(h * 56 + wcol) * CO;
#pragma unroll
        for (int i = 0; i < 8; ++i) {
            const int c = co_w + mbase + i;
            if (c < CO) yb[c] = acc[i] + BIAS[c];
        }
    }
}

// ---------------------------------------------------------------------------
// Anchors: torchvision AnchorGenerator math, stride 4, sizes (16,32,64),
// ratios (0.5,1,2). jnp.round == round-half-even == rintf.
// ---------------------------------------------------------------------------
__global__ void anchors_kernel(float* __restrict__ out) {
    const int row = blockIdx.x * 256 + threadIdx.x;
    if (row >= 28224) return;                        // 56*56*9
    const int p = row / 9, a = row - p * 9;
    const int h = p / 56, w = p - h * 56;
    const int ri = a / 3, si = a - ri * 3;
    const float sizes[3]  = {16.0f, 32.0f, 64.0f};
    const float ratios[3] = {0.5f, 1.0f, 2.0f};
    const float hr = sqrtf(ratios[ri]);
    const float wr = 1.0f / hr;
    const float wsz = wr * sizes[si];
    const float hsz = hr * sizes[si];
    const float gx = (float)w * 4.0f, gy = (float)h * 4.0f;
    float4 v;
    v.x = gx + rintf(-wsz * 0.5f);
    v.y = gy + rintf(-hsz * 0.5f);
    v.z = gx + rintf( wsz * 0.5f);
    v.w = gy + rintf( hsz * 0.5f);
    *(float4*)(out + (size_t)row * 4) = v;
}

// ---------------------------------------------------------------------------
// Launch
// ---------------------------------------------------------------------------
extern "C" void kernel_launch(void* const* d_in, const int* in_sizes, int n_in,
                              void* d_out, int out_size, void* d_ws, size_t ws_size,
                              hipStream_t stream) {
    (void)in_sizes; (void)n_in; (void)out_size; (void)ws_size;
    const float* fm0    = (const float*)d_in[0];
    const float* fm1    = (const float*)d_in[1];
    const float* fm2    = (const float*)d_in[2];
    const float* fm3    = (const float*)d_in[3];
    const float* fuse_w = (const float*)d_in[4];
    const float* fuse_b = (const float*)d_in[5];
    const float* cls_w  = (const float*)d_in[6];
    const float* cls_b  = (const float*)d_in[7];
    const float* cls_hw = (const float*)d_in[8];
    const float* cls_hb = (const float*)d_in[9];
    const float* reg_w  = (const float*)d_in[10];
    const float* reg_b  = (const float*)d_in[11];
    const float* reg_hw = (const float*)d_in[12];
    const float* reg_hb = (const float*)d_in[13];
    float* out = (float*)d_out;

    // workspace carve-up (bf16, 256B aligned chunks) ~122 MB total
    char* ws = (char*)d_ws;
    size_t off = 0;
    auto alloc_bf = [&](size_t elems) -> __bf16* {
        __bf16* p = (__bf16*)(ws + off);
        off += ((elems * 2 + 255) & ~(size_t)255);
        return p;
    };
    const size_t ACT = (size_t)16 * 3136 * 256;
    __bf16* wf   = alloc_bf((size_t)256 * 1024);
    __bf16* wc   = alloc_bf((size_t)1024 * 9 * 256);
    __bf16* wr   = alloc_bf((size_t)1024 * 9 * 256);
    __bf16* wch  = alloc_bf((size_t)96 * 9 * 256);
    __bf16* wrh  = alloc_bf((size_t)48 * 9 * 256);
    __bf16* l0   = alloc_bf(ACT);
    __bf16* l1   = alloc_bf((size_t)16 * 784 * 256);
    __bf16* l2   = alloc_bf((size_t)16 * 196 * 256);
    __bf16* l3   = alloc_bf((size_t)16 * 49 * 256);
    __bf16* bufA = alloc_bf(ACT);
    __bf16* bufB = alloc_bf(ACT);
    __bf16* bufC = alloc_bf(ACT);

    // 1) weight conversion/transposition
    cvt_flat_bf16<<<(256 * 1024 + 255) / 256, 256, 0, stream>>>(fuse_w, wf, 256 * 1024);
    cvt_w3x3_bf16<<<(1024 * 9 * 256 + 255) / 256, 256, 0, stream>>>(cls_w, wc, 1024, 1024);
    cvt_w3x3_bf16<<<(1024 * 9 * 256 + 255) / 256, 256, 0, stream>>>(reg_w, wr, 1024, 1024);
    cvt_w3x3_bf16<<<(96 * 9 * 256 + 255) / 256, 256, 0, stream>>>(cls_hw, wch, 90, 96);
    cvt_w3x3_bf16<<<(48 * 9 * 256 + 255) / 256, 256, 0, stream>>>(reg_hw, wrh, 36, 48);

    // 2) feature maps -> NHWC bf16 at native resolution
    cvt_nchw_nhwc<<<(int)((ACT + 255) / 256), 256, 0, stream>>>(fm0, l0, 3136);
    cvt_nchw_nhwc<<<(16 * 784 * 256 + 255) / 256, 256, 0, stream>>>(fm1, l1, 784);
    cvt_nchw_nhwc<<<(16 * 196 * 256 + 255) / 256, 256, 0, stream>>>(fm2, l2, 196);
    cvt_nchw_nhwc<<<(16 * 49 * 256 + 255) / 256, 256, 0, stream>>>(fm3, l3, 49);

    const dim3 blk(256);
    const dim3 grd_fuse(8, 56, 16);   // (4 co-tiles x 2 w-tiles), h, n
    const dim3 grd_conv(2, 56, 16);   // 2 co-blocks of 128, h, n

    // 3) fused feature map (upsample + concat + 1x1 conv), K = 1024
    fuse_wmma<<<grd_fuse, blk, 0, stream>>>(l0, l1, l2, l3, wf, fuse_b, bufA);

    const size_t WL = (size_t)256 * 9 * 256;   // per-layer subnet weight slice

    // 4) cls subnet (4x conv3x3+ReLU) + cls head
    conv3x3_wmma<<<grd_conv, blk, 0, stream>>>(bufA, wc + 0 * WL, cls_b + 0,   bufB);
    conv3x3_wmma<<<grd_conv, blk, 0, stream>>>(bufB, wc + 1 * WL, cls_b + 256, bufC);
    conv3x3_wmma<<<grd_conv, blk, 0, stream>>>(bufC, wc + 2 * WL, cls_b + 512, bufB);
    conv3x3_wmma<<<grd_conv, blk, 0, stream>>>(bufB, wc + 3 * WL, cls_b + 768, bufC);
    head3x3_wmma<<<dim3(4, 56, 16), blk, 0, stream>>>(bufC, wch, cls_hb, out, 90, 96);

    // 5) reg subnet + reg head
    conv3x3_wmma<<<grd_conv, blk, 0, stream>>>(bufA, wr + 0 * WL, reg_b + 0,   bufB);
    conv3x3_wmma<<<grd_conv, blk, 0, stream>>>(bufB, wr + 1 * WL, reg_b + 256, bufC);
    conv3x3_wmma<<<grd_conv, blk, 0, stream>>>(bufC, wr + 2 * WL, reg_b + 512, bufB);
    conv3x3_wmma<<<grd_conv, blk, 0, stream>>>(bufB, wr + 3 * WL, reg_b + 768, bufC);
    float* reg_out = out + (size_t)16 * 3136 * 90;
    head3x3_wmma<<<dim3(2, 56, 16), blk, 0, stream>>>(bufC, wrh, reg_hb, reg_out, 36, 48);

    // 6) anchors
    float* anch_out = reg_out + (size_t)16 * 3136 * 36;
    anchors_kernel<<<(28224 + 255) / 256, 256, 0, stream>>>(anch_out);
}
